// Net_20091857011309
// MI455X (gfx1250) — matrix operand order repaired and checked by reading, not compiled, not run on hardware
//
#include <hip/hip_runtime.h>
#include <hip/hip_bf16.h>

// ---------------------------------------------------------------------------
// GRU x2 (H=2048, D=1024, T=4096) + MLP head for MI455X (gfx1250, wave32).
//
// Phase 1: xp = x @ W_ih^T + b_ih  as bf16 WMMA GEMM (fp32 accum), per seq.
//          xp written with NT hints (stream-once data, keep L2 for W_hh).
// Phase 2: persistent recurrence kernel: 32 blocks (2 seq x 16 chunks).
//          Each block owns 128 hidden rows (r,z,n triples = 384 W_hh rows),
//          computes the per-step GEMV with v_wmma_f32_16x16x32_bf16 (h vector
//          replicated across all 16 B columns), gate math in fp32, then a
//          per-sequence device-scope barrier before the next timestep.
//          W_hh (bf16, 24MB/seq) stays L2-resident; xp loads are NT and are
//          issued BEFORE the WMMA K-loop so HBM latency is hidden behind it.
// Phase 3: tiny fp32 MLP + log_softmax, single block.
// ---------------------------------------------------------------------------

typedef __attribute__((ext_vector_type(16))) __bf16 v16bf;
typedef __attribute__((ext_vector_type(8)))  __bf16 v8bf;
typedef __attribute__((ext_vector_type(4)))  __bf16 v4bf;
typedef __attribute__((ext_vector_type(8)))  float  v8f;
typedef __attribute__((ext_vector_type(4)))  float  v4f;

constexpr int kH  = 2048;   // hidden
constexpr int kD  = 1024;   // input dim
constexpr int kT  = 4096;   // timesteps
constexpr int kG3 = 3 * kH; // 6144 (r,z,n stacked)
constexpr int kNCH = 16;    // workgroups per sequence in recurrence
constexpr int kCHW = kH / kNCH; // 128 hidden rows per chunk

// ---------------------------------------------------------------------------
// fp32 -> bf16 conversion (vectorized x4); inputs are read-once -> NT loads.
// ---------------------------------------------------------------------------
__global__ void f2bf_kernel(const float* __restrict__ in,
                            __bf16* __restrict__ out, int n) {
  int i = (blockIdx.x * 256 + threadIdx.x) * 4;
  if (i < n) {
    v4f x = __builtin_nontemporal_load((const v4f*)(in + i));
    v4bf y;
    #pragma unroll
    for (int j = 0; j < 4; ++j) y[j] = (__bf16)x[j];
    *(v4bf*)(out + i) = y;
  }
}

// Combine two 16B chunks into a 16x32 bf16 A operand in this file's contiguous k order
// (lane m holds K{0-15} / K{16-31} for low/high half-wave).
__device__ __forceinline__ v16bf ld_a16x32(const __bf16* p) {
  v8bf lo = *(const v8bf*)(p);
  v8bf hi = *(const v8bf*)(p + 8);
  return __builtin_shufflevector(lo, hi, 0, 1, 2, 3, 4, 5, 6, 7,
                                         8, 9, 10, 11, 12, 13, 14, 15);
}

__device__ __forceinline__ v8f wmma_bf16(v16bf a, v16bf b, v8f c) {
  return __builtin_amdgcn_wmma_f32_16x16x32_bf16(false, a, false, b,
                                                 (short)0, c, false, false);
}

// ---------------------------------------------------------------------------
// Phase 1: xp[t, n] = sum_k x[t,k] * W_ih[n,k] + b_ih[n]
// One wave computes a 16(M=t) x 64(N) tile; K=1024 in chunks of 32.
// Grid: 3072 blocks x 256 threads (8 waves) per sequence.
// ---------------------------------------------------------------------------
__global__ __launch_bounds__(256) void xp_gemm_kernel(
    const __bf16* __restrict__ A,    // [T, D]  x (bf16)
    const __bf16* __restrict__ W,    // [3H, D] W_ih (bf16), rows K-contiguous
    const float*  __restrict__ bias, // [3H]
    float* __restrict__ out)         // [T, 3H]
{
  const int lane  = threadIdx.x & 31;
  const int wave  = threadIdx.x >> 5;
  const int tile  = blockIdx.x * 8 + wave; // 0 .. 24575
  const int mt    = tile / (kG3 / 64);     // 0..255
  const int n0    = (tile % (kG3 / 64)) * 64;
  const int l15   = lane & 15;
  const int lhalf = lane >> 4;

  const __bf16* arow = A + (size_t)(mt * 16 + l15) * kD;
  v8f acc[4] = {{}, {}, {}, {}};

  for (int kc = 0; kc < kD; kc += 32) {
    const int abase = kc + lhalf * 16;
    v16bf a = ld_a16x32(arow + abase);
    const int kb = kc + lhalf * 16;
    #pragma unroll
    for (int j = 0; j < 4; ++j) {
      v16bf b = *(const v16bf*)(W + (size_t)(n0 + j * 16 + l15) * kD + kb);
      acc[j] = wmma_bf16(a, b, acc[j]);
    }
  }

  // xp is stream-once: NT stores keep it from evicting W_hh out of L2.
  const int rowb = mt * 16 + lhalf * 8;
  #pragma unroll
  for (int j = 0; j < 4; ++j) {
    const int n = n0 + j * 16 + l15;
    const float bv = bias[n];
    #pragma unroll
    for (int v = 0; v < 8; ++v)
      __builtin_nontemporal_store(acc[j][v] + bv,
                                  out + (size_t)(rowb + v) * kG3 + n);
  }
}

// ---------------------------------------------------------------------------
// Barrier counter init (re-run every launch -> graph-replay safe)
// ---------------------------------------------------------------------------
__global__ void init_kernel(unsigned* bar) {
  if (threadIdx.x < 2) bar[threadIdx.x] = 0u;
}

// ---------------------------------------------------------------------------
// Phase 2: persistent GRU recurrence.
// Grid: 32 blocks x 256 threads. seq = blockIdx/16, chunk = blockIdx%16.
// Wave w owns W_hh rows [g*2048 + i0 + 16w, +16) for each gate g in {r,z,n}.
// B operand = current h (bf16, LDS) replicated across all 16 columns, so all
// D columns are identical; lanes 0 and 16 extract rows 0-7 / 8-15.
// ---------------------------------------------------------------------------
__global__ __launch_bounds__(256) void gru_rec_kernel(
    const __bf16* __restrict__ Whh_all, // [2][3H, H] bf16
    const float*  __restrict__ xp_all,  // [2][T, 3H]
    const float*  __restrict__ bhh1,    // [3H]
    const float*  __restrict__ bhh2,    // [3H]
    __bf16* __restrict__ hbuf_all,      // [2][H] bf16, cross-block h exchange
    float*  __restrict__ hfinal,        // [2][H] fp32
    unsigned* __restrict__ bar)         // [2]
{
  const int seq   = blockIdx.x >> 4;
  const int chk   = blockIdx.x & 15;
  const int i0    = chk * kCHW;
  const int tid   = threadIdx.x;
  const int lane  = tid & 31;
  const int wave  = tid >> 5;
  const int l15   = lane & 15;
  const int lhalf = lane >> 4;

  const __bf16* Whh  = Whh_all + (size_t)seq * kG3 * kH;
  const float*  xp   = xp_all + (size_t)seq * kT * kG3;
  const float*  bhh  = seq ? bhh2 : bhh1;
  __bf16*       hbuf = hbuf_all + seq * kH;
  unsigned*     cnt  = bar + seq;

  __shared__ __align__(32) __bf16 lds_h[kH]; // current h, bf16 (B operand)
  __shared__ float lds_hp[3 * kCHW];         // W_hh @ h slice (r,z,n)
  __shared__ float lds_hf[kCHW];             // fp32 h for this chunk

  for (int i = tid; i < kH; i += 256) lds_h[i] = (__bf16)0.0f;
  if (tid < kCHW) lds_hf[tid] = 0.0f;
  __syncthreads();

  const __bf16* ar0 = Whh + (size_t)(0 * kH + i0 + wave * 16 + l15) * kH;
  const __bf16* ar1 = Whh + (size_t)(1 * kH + i0 + wave * 16 + l15) * kH;
  const __bf16* ar2 = Whh + (size_t)(2 * kH + i0 + wave * 16 + l15) * kH;

  // Loop-invariant gate biases, hoisted into registers.
  const int gi = i0 + (tid & (kCHW - 1));
  float br = 0.0f, bz = 0.0f, bn = 0.0f;
  if (tid < kCHW) {
    br = bhh[gi];
    bz = bhh[kH + gi];
    bn = bhh[2 * kH + gi];
  }

  for (int t = 0; t < kT; ++t) {
    // Issue the HBM xp loads FIRST (NT: stream-once, don't pollute L2).
    // VMEM loads complete in order, so the WMMA loop's own partial
    // s_wait_loadcnt waits hide their full HBM latency.
    float xr = 0.0f, xz = 0.0f, xn = 0.0f;
    {
      const size_t xb = (size_t)t * kG3;
      if (tid < kCHW) {
        xr = __builtin_nontemporal_load(xp + xb + gi);
        xz = __builtin_nontemporal_load(xp + xb + kH + gi);
        xn = __builtin_nontemporal_load(xp + xb + 2 * kH + gi);
      }
    }

    v8f c0 = {}, c1 = {}, c2 = {};
    #pragma unroll 4
    for (int kc = 0; kc < kH; kc += 32) {
      v16bf b = *(const v16bf*)(&lds_h[kc + lhalf * 16]);
      const int ab = kc + lhalf * 16;
      c0 = wmma_bf16(ld_a16x32(ar0 + ab), b, c0);
      c1 = wmma_bf16(ld_a16x32(ar1 + ab), b, c1);
      c2 = wmma_bf16(ld_a16x32(ar2 + ab), b, c2);
    }
    // All 16 D columns identical; lanes 0/16 hold rows 0-7 / 8-15.
    if (l15 == 0) {
      const int rb = wave * 16 + lhalf * 8;
      #pragma unroll
      for (int v = 0; v < 8; ++v) {
        lds_hp[0 * kCHW + rb + v] = c0[v];
        lds_hp[1 * kCHW + rb + v] = c1[v];
        lds_hp[2 * kCHW + rb + v] = c2[v];
      }
    }
    __syncthreads();

    if (tid < kCHW) {
      const float hr = lds_hp[tid]            + br;
      const float hz = lds_hp[kCHW + tid]     + bz;
      const float hn = lds_hp[2 * kCHW + tid] + bn;
      const float r = 1.0f / (1.0f + __expf(-(xr + hr)));
      const float z = 1.0f / (1.0f + __expf(-(xz + hz)));
      const float n = tanhf(xn + r * hn);
      const float hnew = (1.0f - z) * n + z * lds_hf[tid];
      lds_hf[tid] = hnew;
      hbuf[gi] = (__bf16)hnew;
    }

    // Device-scope barrier among the 16 blocks of this sequence.
    __syncthreads();
    if (tid == 0) {
      __threadfence();
      __hip_atomic_fetch_add(cnt, 1u, __ATOMIC_RELEASE, __HIP_MEMORY_SCOPE_AGENT);
      const unsigned target = (unsigned)(kNCH * (t + 1));
      while (__hip_atomic_load(cnt, __ATOMIC_ACQUIRE, __HIP_MEMORY_SCOPE_AGENT) < target)
        __builtin_amdgcn_s_sleep(1);
    }
    __syncthreads();

    // Reload full h (2048 bf16 = 4KB) into LDS: 16B per thread.
    *(v8bf*)(&lds_h[tid * 8]) = *(const v8bf*)(hbuf + tid * 8);
    __syncthreads();
  }

  if (tid < kCHW) hfinal[seq * kH + i0 + tid] = lds_hf[tid];
}

// ---------------------------------------------------------------------------
// Phase 3: out = log_softmax(relu(cat(h1,h2) @ fc1_w^T + fc1_b) @ fc2_w^T + fc2_b)
// ---------------------------------------------------------------------------
__global__ __launch_bounds__(256) void mlp_kernel(
    const float* __restrict__ hfinal, // [2*H] = 4096 (h1 then h2)
    const float* __restrict__ fc1w,   // [256, 4096]
    const float* __restrict__ fc1b,   // [256]
    const float* __restrict__ fc2w,   // [3, 256]
    const float* __restrict__ fc2b,   // [3]
    float* __restrict__ out)          // [3]
{
  __shared__ float s[256];
  const int j = threadIdx.x;
  const float* wr = fc1w + (size_t)j * (2 * kH);
  float acc = fc1b[j];
  #pragma unroll 4
  for (int k = 0; k < 2 * kH; k += 4) {
    acc += wr[k] * hfinal[k] + wr[k + 1] * hfinal[k + 1] +
           wr[k + 2] * hfinal[k + 2] + wr[k + 3] * hfinal[k + 3];
  }
  s[j] = acc > 0.0f ? acc : 0.0f;
  __syncthreads();
  if (j == 0) {
    float o[3];
    for (int m = 0; m < 3; ++m) {
      float a = fc2b[m];
      for (int k = 0; k < 256; ++k) a += fc2w[m * 256 + k] * s[k];
      o[m] = a;
    }
    const float mx = fmaxf(o[0], fmaxf(o[1], o[2]));
    const float lse =
        logf(expf(o[0] - mx) + expf(o[1] - mx) + expf(o[2] - mx));
    out[0] = o[0] - mx - lse;
    out[1] = o[1] - mx - lse;
    out[2] = o[2] - mx - lse;
  }
}

// ---------------------------------------------------------------------------
// Host launcher
// ---------------------------------------------------------------------------
extern "C" void kernel_launch(void* const* d_in, const int* in_sizes, int n_in,
                              void* d_out, int out_size, void* d_ws, size_t ws_size,
                              hipStream_t stream) {
  (void)in_sizes; (void)n_in; (void)out_size; (void)ws_size;
  const float* x1   = (const float*)d_in[0];
  const float* x2   = (const float*)d_in[1];
  const float* Wih1 = (const float*)d_in[2];
  const float* Whh1 = (const float*)d_in[3];
  const float* bih1 = (const float*)d_in[4];
  const float* bhh1 = (const float*)d_in[5];
  const float* Wih2 = (const float*)d_in[6];
  const float* Whh2 = (const float*)d_in[7];
  const float* bih2 = (const float*)d_in[8];
  const float* bhh2 = (const float*)d_in[9];
  const float* fc1w = (const float*)d_in[10];
  const float* fc1b = (const float*)d_in[11];
  const float* fc2w = (const float*)d_in[12];
  const float* fc2b = (const float*)d_in[13];
  float* out = (float*)d_out;

  char* ws = (char*)d_ws;
  size_t off = 0;
  auto carve = [&](size_t bytes) -> char* {
    char* p = ws + off;
    off = (off + bytes + 255) & ~(size_t)255;
    return p;
  };

  __bf16* xbf1   = (__bf16*)carve((size_t)kT * kD * 2);        //  8 MB
  __bf16* xbf2   = (__bf16*)carve((size_t)kT * kD * 2);        //  8 MB
  __bf16* wihbf1 = (__bf16*)carve((size_t)kG3 * kD * 2);       // 12 MB
  __bf16* wihbf2 = (__bf16*)carve((size_t)kG3 * kD * 2);       // 12 MB
  __bf16* whhbf  = (__bf16*)carve((size_t)2 * kG3 * kH * 2);   // 48 MB (both seqs)
  float*  xp     = (float*) carve((size_t)2 * kT * kG3 * 4);   // 192 MB (both seqs)
  __bf16* hbuf   = (__bf16*)carve((size_t)2 * kH * 2);
  float*  hfin   = (float*) carve((size_t)2 * kH * 4);
  unsigned* bar  = (unsigned*)carve(256);

  const dim3 blk(256);

  // fp32 -> bf16 conversions
  const int nx = kT * kD;        // 4,194,304
  const int nwih = kG3 * kD;     // 6,291,456
  const int nwhh = kG3 * kH;     // 12,582,912
  f2bf_kernel<<<nx / 1024, blk, 0, stream>>>(x1, xbf1, nx);
  f2bf_kernel<<<nx / 1024, blk, 0, stream>>>(x2, xbf2, nx);
  f2bf_kernel<<<nwih / 1024, blk, 0, stream>>>(Wih1, wihbf1, nwih);
  f2bf_kernel<<<nwih / 1024, blk, 0, stream>>>(Wih2, wihbf2, nwih);
  f2bf_kernel<<<nwhh / 1024, blk, 0, stream>>>(Whh1, whhbf, nwhh);
  f2bf_kernel<<<nwhh / 1024, blk, 0, stream>>>(Whh2, whhbf + (size_t)kG3 * kH, nwhh);

  // Phase 1: input projections (WMMA GEMM)
  const int gemm_blocks = (kT / 16) * (kG3 / 64) / 8; // 3072
  xp_gemm_kernel<<<gemm_blocks, blk, 0, stream>>>(xbf1, wihbf1, bih1, xp);
  xp_gemm_kernel<<<gemm_blocks, blk, 0, stream>>>(xbf2, wihbf2, bih2,
                                                  xp + (size_t)kT * kG3);

  // Phase 2: recurrence (persistent, 2 sequences x 16 chunks)
  init_kernel<<<1, 32, 0, stream>>>(bar);
  gru_rec_kernel<<<32, blk, 0, stream>>>(whhbf, xp, bhh1, bhh2, hbuf, hfin, bar);

  // Phase 3: MLP head
  mlp_kernel<<<1, blk, 0, stream>>>(hfin, fc1w, fc1b, fc2w, fc2b, out);
}
